// CMCM_38817914421377
// MI455X (gfx1250) — compile-verified, matrix operand order checked by reading, not run on hardware
//
#include <hip/hip_runtime.h>
#include <hip/hip_bf16.h>
#include <math.h>

typedef __attribute__((ext_vector_type(16))) _Float16 v16h;
typedef __attribute__((ext_vector_type(8)))  _Float16 v8h;
typedef __attribute__((ext_vector_type(8)))  float    v8f;

#define NB    8
#define NC    19
#define NH    320
#define NW    2048
#define NP    1280          // 10 block-rows * 128 block-cols
#define BLKW  128           // block cols per batch
#define EPB   ((size_t)NP * NP)   // elements of e per batch

// ---------------------------------------------------------------------------
// Kernel 1: 16x16 block-sum over pixels, argmax over 19 channels, emit one-hot
// f16 rows [NB*NP][32] into workspace. One wave per pooling block.
// Only block-rows 10..19 (pixel rows 160..319) are needed.
// ---------------------------------------------------------------------------
__global__ void pool_argmax_onehot(const float* __restrict__ label,
                                   _Float16* __restrict__ onehot) {
    const int wave = threadIdx.x >> 5;           // 8 waves / block
    const int lane = threadIdx.x & 31;
    const int id   = blockIdx.x * 8 + wave;      // 0 .. 10239
    const int b    = id / NP;
    const int rem  = id % NP;
    const int bi   = 10 + rem / BLKW;            // block row 10..19
    const int bj   = rem % BLKW;                 // block col 0..127

    const int r0 = lane >> 2;                    // 0..7
    const int cg = lane & 3;                     // float4 group
    const size_t ch_stride = (size_t)NH * NW;
    const size_t base = (size_t)b * NC * ch_stride
                      + (size_t)(16 * bi + r0) * NW
                      + (size_t)(16 * bj + 4 * cg);

    float bestv = -INFINITY;
    int   bestc = 0;
    const float* p = label + base;
    for (int c = 0; c < NC; ++c) {
        // prefetch next channel's data for this lane
        if (c + 1 < NC) __builtin_prefetch(p + ch_stride, 0, 1);
        float4 lo = *(const float4*)(p);                    // row r0
        float4 hi = *(const float4*)(p + (size_t)8 * NW);   // row r0+8
        float s = lo.x + lo.y + lo.z + lo.w + hi.x + hi.y + hi.z + hi.w;
        // full wave32 sum
        #pragma unroll
        for (int off = 16; off >= 1; off >>= 1)
            s += __shfl_xor(s, off, 32);
        if (s > bestv) { bestv = s; bestc = c; }            // first-max wins
        p += ch_stride;
    }

    // 32 lanes write one f16 each -> 64B coalesced one-hot row (19 real + pad)
    const int prow = (bi - 10) * BLKW + bj;
    _Float16* orow = onehot + ((size_t)b * NP + prow) * 32;
    orow[lane] = (lane == bestc) ? (_Float16)1.0f : (_Float16)0.0f;
}

// ---------------------------------------------------------------------------
// Fragment loader: ISA 16-bit A-matrix 16x32 layout (B 32x16 is its mirror):
//   lanes 0-15  : K = 0..7  and 16..23 of row (lane)
//   lanes 16-31 : K = 8..15 and 24..31 of row (lane-16)
// One-hot row is 32 halves (64B, 16B-aligned chunks).
// ---------------------------------------------------------------------------
__device__ inline v16h load_oh_frag(const _Float16* __restrict__ oh,
                                    int row, int lane) {
    const _Float16* rp = oh + (size_t)row * 32;
    const int kb = (lane < 16) ? 0 : 8;
    v8h lo = *(const v8h*)(rp + kb);        // K = kb .. kb+7
    v8h hi = *(const v8h*)(rp + kb + 16);   // K = kb+16 .. kb+23
    v16h f;
    #pragma unroll
    for (int t = 0; t < 8; ++t) { f[t] = lo[t]; f[8 + t] = hi[t]; }
    return f;
}

// ---------------------------------------------------------------------------
// Kernel 2: per (batch, 16-row tile): same-mask via WMMA onehot x onehot^T,
// rewrite e from energy, stash 16x1280 e tile in LDS, row-softmax, write both.
// 128 threads = 4 waves; wave w handles column tiles w, w+4, ... (20 each).
// ---------------------------------------------------------------------------
__global__ void mask_softmax(const float* __restrict__ energy,
                             const _Float16* __restrict__ onehot,
                             float* __restrict__ e_out,
                             float* __restrict__ att_out) {
    extern __shared__ float smem[];
    float* e_s  = smem;                 // 16*1280
    float* red  = smem + 16 * NP;       // 16*8 partials
    float* rowm = red + 128;            // 16 row maxes
    float* rowsum = rowm + 16;          // 16 row sums

    const int tid  = threadIdx.x;
    const int wave = tid >> 5;
    const int lane = tid & 31;
    const int b    = blockIdx.x / 80;
    const int pi   = blockIdx.x % 80;   // row tile
    const int row0 = pi * 16;

    const _Float16* ohb = onehot + (size_t)b * NP * 32;
    const float*    enb = energy + (size_t)b * EPB;
    float*          eob = e_out  + (size_t)b * EPB;

    // A fragment: rows row0..row0+15 (constant for all column tiles)
    const v16h afrag = load_oh_frag(ohb, row0 + (lane & 15), lane);

    const int n     = lane & 15;                  // D column within tile
    const int mbase = (lane < 16) ? 0 : 8;        // D row base (VGPR v -> row mbase+v)

    for (int t = wave; t < 80; t += 4) {
        const v16h bfrag = load_oh_frag(ohb, t * 16 + (lane & 15), lane);
        v8f acc = {};
        acc = __builtin_amdgcn_wmma_f32_16x16x32_f16(
                  false, afrag, false, bfrag, (short)0, acc, false, false);

        const int col = t * 16 + n;
        const size_t gbase = (size_t)(row0 + mbase) * NP + col;
        #pragma unroll
        for (int v = 0; v < 8; ++v) {
            const size_t gidx = gbase + (size_t)v * NP;
            const float  en   = enb[gidx];
            const bool   same = acc[v] > 0.5f;
            const float  ev   = same ? (en < 0.0f ?  0.5f : en)
                                     : (en > 0.0f ? -0.5f : en);
            eob[gidx] = ev;
            e_s[(mbase + v) * NP + col] = ev;
        }
    }
    __syncthreads();

    // Row softmax over 1280 cols: 8 threads per row, 160 cols each.
    const int r   = tid >> 3;           // 0..15
    const int sub = tid & 7;
    const int c0  = sub * 160;
    float* erow = e_s + r * NP;

    float mx = -INFINITY;
    for (int q = 0; q < 160; ++q) mx = fmaxf(mx, erow[c0 + q]);
    red[r * 8 + sub] = mx;
    __syncthreads();
    if (sub == 0) {
        float m = red[r * 8];
        #pragma unroll
        for (int k = 1; k < 8; ++k) m = fmaxf(m, red[r * 8 + k]);
        rowm[r] = m;
    }
    __syncthreads();

    const float m = rowm[r];
    float s = 0.0f;
    for (int q = 0; q < 160; ++q) s += __expf(erow[c0 + q] - m);
    __syncthreads();          // red[] reuse
    red[r * 8 + sub] = s;
    __syncthreads();
    if (sub == 0) {
        float t = red[r * 8];
        #pragma unroll
        for (int k = 1; k < 8; ++k) t += red[r * 8 + k];
        rowsum[r] = t;
    }
    __syncthreads();

    const float inv = 1.0f / rowsum[r];
    float* arow = att_out + (size_t)b * EPB + (size_t)(row0 + r) * NP;
    for (int q = 0; q < 160; ++q)
        arow[c0 + q] = __expf(erow[c0 + q] - m) * inv;
}

// ---------------------------------------------------------------------------
extern "C" void kernel_launch(void* const* d_in, const int* in_sizes, int n_in,
                              void* d_out, int out_size, void* d_ws, size_t ws_size,
                              hipStream_t stream) {
    (void)in_sizes; (void)n_in; (void)out_size; (void)ws_size;
    const float* label  = (const float*)d_in[0];
    const float* energy = (const float*)d_in[1];
    float* e_out   = (float*)d_out;
    float* att_out = (float*)d_out + (size_t)NB * EPB;
    _Float16* onehot = (_Float16*)d_ws;   // NB*NP*32 halves = 640 KB

    // K1: 8*10*128 = 10240 pooling blocks, 8 waves per 256-thread block
    pool_argmax_onehot<<<1280, 256, 0, stream>>>(label, onehot);

    // K2: 8 batches * 80 row-tiles, 128 threads, LDS: 16*1280 + 128 + 32 floats
    const size_t smem = (size_t)(16 * NP + 128 + 32) * sizeof(float);
    mask_softmax<<<NB * 80, 128, smem, stream>>>(energy, onehot, e_out, att_out);
}
